// SparseAttention_75703093559829
// MI455X (gfx1250) — matrix operand order repaired and checked by reading, not compiled, not run on hardware
//
#include <hip/hip_runtime.h>

typedef __attribute__((ext_vector_type(16))) _Float16 v16h;
typedef __attribute__((ext_vector_type(8)))  float    v8f;

#define SEQ_T 4096
#define DIM_E 512
#define BM 64            // q rows per workgroup
#define BN 64            // key rows per tile
#define QSTR 520         // LDS row stride (halves) for Q and K tiles
#define VSTR 68          // LDS row stride (halves) for transposed V
#define PSTR 72          // LDS row stride (halves) for P
#define NKT (SEQ_T / BN) // 64 key tiles

// LDS layout (halves): sQ[BM][QSTR] | sK[BN][QSTR] | sV[DIM_E][VSTR] | sP[BM][PSTR] | stats(512 f32)
#define SMEM_HALVES ((size_t)(BM*QSTR + BN*QSTR + DIM_E*VSTR + BM*PSTR))
#define SMEM_BYTES  (SMEM_HALVES*2 + 8*BM*4)

__global__ __launch_bounds__(256) void
sparse_attn_flash_wmma(const float* __restrict__ qg_, const float* __restrict__ kg_,
                       const float* __restrict__ vg_, float* __restrict__ og_)
{
    extern __shared__ char smem_raw[];
    _Float16* sQ = (_Float16*)smem_raw;
    _Float16* sK = sQ + BM*QSTR;
    _Float16* sV = sK + BN*QSTR;
    _Float16* sP = sV + DIM_E*VSTR;
    float* stats = (float*)(sP + BM*PSTR);
    float* m_run = stats;            // [BM]
    float* l_run = m_run + BM;       // [BM]
    float* a_s   = l_run + BM;       // [BM] alpha for current tile
    float* mn_s  = a_s   + BM;       // [BM] new running max
    float* pmax  = mn_s  + BM;       // [BM][2]
    float* psum  = pmax  + 2*BM;     // [BM][2]

    const int tid  = threadIdx.x;
    const int lane = tid & 31;
    const int w    = tid >> 5;       // 0..7
    const int mi   = w >> 1;         // 0..3  : q row-tile
    const int jp   = w & 1;          // 0..1  : column group
    const int row16 = lane & 15;
    const int halfSel = lane >> 4;           // 0 or 1
    const int aKoff = halfSel ? 16 : 0;      // A-matrix K offset per lane-half
    const int bKoff = halfSel ? 16 : 0;      // B-matrix K offset per lane-half

    const int b  = blockIdx.y;
    const int q0 = blockIdx.x * BM;

    // ---- stage Q tile (pre-scaled by 1/sqrt(512/8) = 0.125), f32 -> f16 LDS ----
    {
        const float* qg = qg_ + ((size_t)b * SEQ_T + q0) * DIM_E;
        for (int i = tid; i < BM*DIM_E/4; i += 256) {
            int off = i << 2;
            int r = off >> 9, c = off & 511;
            float4 f = *(const float4*)(qg + off);
            union { _Float16 h[4]; uint2 u2; } pk;
            pk.h[0] = (_Float16)(f.x * 0.125f);
            pk.h[1] = (_Float16)(f.y * 0.125f);
            pk.h[2] = (_Float16)(f.z * 0.125f);
            pk.h[3] = (_Float16)(f.w * 0.125f);
            *(uint2*)(sQ + r*QSTR + c) = pk.u2;
        }
        for (int i = tid; i < BM; i += 256) { m_run[i] = -3.0e38f; l_run[i] = 0.0f; }
    }

    v8f o[16];
    v8f vzero = {};
    #pragma unroll
    for (int t = 0; t < 16; ++t) o[t] = vzero;

    for (int kt = 0; kt < NKT; ++kt) {
        __syncthreads();  // previous iteration done reading sK/sV/sP
        // ---- stage K tile (row-major) and V tile (transposed) ----
        {
            const float* kg = kg_ + ((size_t)b * SEQ_T + (size_t)kt * BN) * DIM_E;
            const float* vg = vg_ + ((size_t)b * SEQ_T + (size_t)kt * BN) * DIM_E;
            for (int i = tid; i < BN*DIM_E/4; i += 256) {
                int off = i << 2;
                int r = off >> 9, c = off & 511;
                float4 f = *(const float4*)(kg + off);
                union { _Float16 h[4]; uint2 u2; } pk;
                pk.h[0] = (_Float16)f.x; pk.h[1] = (_Float16)f.y;
                pk.h[2] = (_Float16)f.z; pk.h[3] = (_Float16)f.w;
                *(uint2*)(sK + r*QSTR + c) = pk.u2;
                float4 g = *(const float4*)(vg + off);
                sV[(c+0)*VSTR + r] = (_Float16)g.x;
                sV[(c+1)*VSTR + r] = (_Float16)g.y;
                sV[(c+2)*VSTR + r] = (_Float16)g.z;
                sV[(c+3)*VSTR + r] = (_Float16)g.w;
            }
        }
        __syncthreads();

        // ---- S = Qs * K^T : two 16x16 tiles per wave (keys 32*jp .. 32*jp+31) ----
        v8f c0 = vzero, c1 = vzero;
        const _Float16* qrow = sQ + (mi*16 + row16) * QSTR;
        const _Float16* krow0 = sK + (32*jp      + row16) * QSTR;
        const _Float16* krow1 = sK + (32*jp + 16 + row16) * QSTR;
        #pragma unroll 2
        for (int d = 0; d < DIM_E; d += 32) {
            v16h a, b0, b1;
            #pragma unroll
            for (int vv = 0; vv < 8; ++vv) {
                int kk = d + aKoff + 2*vv;
                a[2*vv]   = qrow[kk];
                a[2*vv+1] = qrow[kk+1];
            }
            #pragma unroll
            for (int vv = 0; vv < 8; ++vv) {
                int dk = d + bKoff + 2*vv;
                b0[2*vv]   = krow0[dk];
                b0[2*vv+1] = krow0[dk+1];
                b1[2*vv]   = krow1[dk];
                b1[2*vv+1] = krow1[dk+1];
            }
            c0 = __builtin_amdgcn_wmma_f32_16x16x32_f16(false, a, false, b0, (short)0, c0, false, false);
            c1 = __builtin_amdgcn_wmma_f32_16x16x32_f16(false, a, false, b1, (short)0, c1, false, false);
        }

        // ---- per-row partial max over this wave's 32 key columns ----
        float pm[8];
        #pragma unroll
        for (int vv = 0; vv < 8; ++vv) {
            float x = fmaxf(c0[vv], c1[vv]);
            #pragma unroll
            for (int m = 1; m < 16; m <<= 1) x = fmaxf(x, __shfl_xor(x, m, 32));
            pm[vv] = x;
        }
        if (row16 == 0) {
            #pragma unroll
            for (int vv = 0; vv < 8; ++vv)
                pmax[(mi*16 + halfSel*8 + vv)*2 + jp] = pm[vv];
        }
        __syncthreads();

        // ---- online-softmax stats (one wave per row group updates) ----
        if (jp == 0 && lane < 16) {
            int r = mi*16 + lane;
            float mo = m_run[r];
            float tm = fmaxf(pmax[r*2+0], pmax[r*2+1]);
            float mn = fmaxf(mo, tm);
            a_s[r]  = __expf(mo - mn);
            m_run[r] = mn;
            mn_s[r]  = mn;
        }
        __syncthreads();

        // ---- P = exp(S - m), write f16 to LDS, row-sum partials, rescale O ----
        float al8[8], mn8[8];
        #pragma unroll
        for (int vv = 0; vv < 8; ++vv) {
            int r = mi*16 + halfSel*8 + vv;
            al8[vv] = a_s[r]; mn8[vv] = mn_s[r];
        }
        float ps[8];
        #pragma unroll
        for (int vv = 0; vv < 8; ++vv) {
            float p0 = __expf(c0[vv] - mn8[vv]);
            float p1 = __expf(c1[vv] - mn8[vv]);
            int r = mi*16 + halfSel*8 + vv;
            sP[r*PSTR + 32*jp      + row16] = (_Float16)p0;
            sP[r*PSTR + 32*jp + 16 + row16] = (_Float16)p1;
            float s = p0 + p1;
            #pragma unroll
            for (int m = 1; m < 16; m <<= 1) s += __shfl_xor(s, m, 32);
            ps[vv] = s;
        }
        if (row16 == 0) {
            #pragma unroll
            for (int vv = 0; vv < 8; ++vv)
                psum[(mi*16 + halfSel*8 + vv)*2 + jp] = ps[vv];
        }
        #pragma unroll
        for (int t = 0; t < 16; ++t) {
            #pragma unroll
            for (int vv = 0; vv < 8; ++vv) o[t][vv] *= al8[vv];
        }
        __syncthreads();
        if (jp == 0 && lane < 16) {
            int r = mi*16 + lane;
            l_run[r] = a_s[r]*l_run[r] + psum[r*2+0] + psum[r*2+1];
        }

        // ---- O += P * V : 16 column tiles, kdim 0..63 in 2 chunks ----
        v16h a0, a1;
        const _Float16* prow = sP + (mi*16 + row16) * PSTR;
        #pragma unroll
        for (int vv = 0; vv < 8; ++vv) {
            int kk = aKoff + 2*vv;
            a0[2*vv]   = prow[kk];      a0[2*vv+1] = prow[kk+1];
            a1[2*vv]   = prow[32+kk];   a1[2*vv+1] = prow[32+kk+1];
        }
        #pragma unroll
        for (int t = 0; t < 16; ++t) {
            const _Float16* vrow = sV + (jp*256 + t*16 + row16) * VSTR;
            v16h b0, b1;
            #pragma unroll
            for (int vv = 0; vv < 8; ++vv) {
                int key0 = bKoff + 2*vv;
                b0[2*vv]   = vrow[key0];      b0[2*vv+1] = vrow[key0+1];
                b1[2*vv]   = vrow[32+key0];   b1[2*vv+1] = vrow[32+key0+1];
            }
            o[t] = __builtin_amdgcn_wmma_f32_16x16x32_f16(false, a0, false, b0, (short)0, o[t], false, false);
            o[t] = __builtin_amdgcn_wmma_f32_16x16x32_f16(false, a1, false, b1, (short)0, o[t], false, false);
        }
    }

    __syncthreads();
    // ---- finalize: divide by l and store f32 output ----
    float li[8];
    #pragma unroll
    for (int vv = 0; vv < 8; ++vv) {
        int r = mi*16 + halfSel*8 + vv;
        li[vv] = 1.0f / l_run[r];
    }
    float* og = og_ + ((size_t)b * SEQ_T + q0) * DIM_E;
    #pragma unroll
    for (int t = 0; t < 16; ++t) {
        int col = jp*256 + t*16 + row16;
        #pragma unroll
        for (int vv = 0; vv < 8; ++vv) {
            int r = mi*16 + halfSel*8 + vv;
            og[(size_t)r*DIM_E + col] = o[t][vv] * li[vv];
        }
    }
}

extern "C" void kernel_launch(void* const* d_in, const int* in_sizes, int n_in,
                              void* d_out, int out_size, void* d_ws, size_t ws_size,
                              hipStream_t stream) {
    (void)in_sizes; (void)n_in; (void)out_size; (void)d_ws; (void)ws_size;
    const float* q = (const float*)d_in[0];
    const float* k = (const float*)d_in[1];
    const float* v = (const float*)d_in[2];
    float* out = (float*)d_out;
    dim3 grid(SEQ_T / BM, 4, 1);   // 64 q-tiles x 4 batches
    dim3 block(256, 1, 1);
    sparse_attn_flash_wmma<<<grid, block, SMEM_BYTES, stream>>>(q, k, v, out);
}
